// SymbolicTransformerModelWithReconstruction_10969346474690
// MI455X (gfx1250) — compile-verified, matrix operand order checked by reading, not run on hardware
//
#include <hip/hip_runtime.h>
#include <cstdint>
#include <cstddef>

// ---------------- model constants ----------------
#define N_LAYER  4
#define N_HEAD   16
#define N_EMBD   1024
#define HEAD_DIM 64
#define VOCAB    4096
#define BB       2
#define TT       1024
#define BT       (BB * TT)
#define QKV_LD   (3 * N_EMBD)
#define NEG_BIG  (-3.0e38f)

typedef __attribute__((ext_vector_type(8)))  _Float16 v8h;
typedef __attribute__((ext_vector_type(16))) _Float16 v16h;
typedef __attribute__((ext_vector_type(8)))  float    v8f;
typedef __attribute__((ext_vector_type(4)))  unsigned int v4u;
typedef __attribute__((ext_vector_type(4)))  int      v4i;
typedef __attribute__((ext_vector_type(8)))  int      v8i;

#if defined(__has_builtin)
#if __has_builtin(__builtin_amdgcn_tensor_load_to_lds) && \
    __has_builtin(__builtin_amdgcn_s_wait_tensorcnt)
#define HAVE_TDM 1
#endif
#endif
#ifndef HAVE_TDM
#define HAVE_TDM 0
#endif

// ---------------- WMMA helpers (CDNA5 16x16x32 f16 -> f32) ----------------
static __device__ __forceinline__ v8f wmma16(v16h a, v16h b, v8f c) {
  // 8 args: (neg_a, A, neg_b, B, c_mod, C, reuse_a, reuse_b)
  return __builtin_amdgcn_wmma_f32_16x16x32_f16(false, a, false, b, (short)0, c,
                                                false, false);
}

// Fragment element e of lane L maps to K = 16*(e/8) + 8*(L>>4) + (e%8),
// row/col = L&15.  Two contiguous 8xf16 (b128) loads per fragment.
static __device__ __forceinline__ v16h frag_pack(const _Float16* p) {
  v8h lo = *(const v8h*)p;
  v8h hh = *(const v8h*)(p + 16);
  return __builtin_shufflevector(lo, hh, 0, 1, 2, 3, 4, 5, 6, 7, 8, 9, 10, 11,
                                 12, 13, 14, 15);
}

static __device__ __forceinline__ v16h frag_ld_g(const _Float16* __restrict__ base,
                                                 int ld, int row0, int col0) {
  const int lane = threadIdx.x & 31;
  return frag_pack(base + (size_t)(row0 + (lane & 15)) * ld + col0 +
                   ((lane >> 4) << 3));
}

static __device__ __forceinline__ v16h frag_ld_g_clamp(
    const _Float16* __restrict__ base, int ld, int row0, int col0, int rowmax) {
  const int lane = threadIdx.x & 31;
  int r = row0 + (lane & 15);
  if (r > rowmax) r = rowmax;
  return frag_pack(base + (size_t)r * ld + col0 + ((lane >> 4) << 3));
}

// ---------------- Tensor Data Mover: 2D f16 tile (32 rows x 64 cols) -> LDS ----
// D# per CDNA5 ISA:
//  group0: count=1 | lds_addr[63:32] | global_addr[120:64] | type=2 [127:126]
//  group1: data_size=1(2B) [17:16] | tensor_dim0 [79:48] | tensor_dim1 [111:80]
//          tile_dim0 [127:112] | tile_dim1 [143:128] | dim0_stride [207:160]
// OOB rows (>= tensor_dim1) read as zero, which subsumes tail clamping.
static __device__ __forceinline__ void tdm_load_tile_f16(
    unsigned lds_addr, const _Float16* gsrc, unsigned rows_avail,
    unsigned row_stride_elems) {
#if HAVE_TDM
  const unsigned long long ga = (unsigned long long)(uintptr_t)gsrc;
  const unsigned td0 = 64u, tile0 = 64u, tile1 = 32u;
  v4u g0;
  g0[0] = 1u;                                                // count=1
  g0[1] = lds_addr;                                          // lds_addr
  g0[2] = (unsigned)(ga & 0xffffffffu);                      // global_addr lo
  g0[3] = (unsigned)((ga >> 32) & 0x01ffffffu) | (2u << 30); // ga hi | type=2
  v8i g1;
  g1[0] = (int)(1u << 16);                                   // data_size=2B
  g1[1] = (int)((td0 & 0xffffu) << 16);                      // tensor_dim0 lo
  g1[2] = (int)((td0 >> 16) | ((rows_avail & 0xffffu) << 16)); // td0 hi|td1 lo
  g1[3] = (int)((rows_avail >> 16) | (tile0 << 16));         // td1 hi | tile0
  g1[4] = (int)tile1;                                        // tile1, tile2=0
  g1[5] = (int)row_stride_elems;                             // dim0 stride lo
  g1[6] = 0;                                                 // stride hi, s1
  g1[7] = 0;
  const v4i z4 = {0, 0, 0, 0};
#if defined(__clang_major__) && (__clang_major__ >= 23)
  const v8i z8 = {0, 0, 0, 0, 0, 0, 0, 0};
  __builtin_amdgcn_tensor_load_to_lds(g0, g1, z4, z4, z8, 0);
#else
  __builtin_amdgcn_tensor_load_to_lds(g0, g1, z4, z4, 0);
#endif
#else
  (void)lds_addr; (void)gsrc; (void)rows_avail; (void)row_stride_elems;
#endif
}

// Fallback VGPR staging (used only when TDM builtin unavailable)
static __device__ __forceinline__ void stage_tile_vgpr(
    _Float16* Vl, const _Float16* gsrc, int rows_avail, int row_stride_elems) {
  const int lane = threadIdx.x & 31;
  int vr = lane;
  if (vr > rows_avail - 1) vr = rows_avail - 1;
  const _Float16* src = gsrc + (size_t)vr * row_stride_elems;
  v8h* dst = (v8h*)(Vl + lane * 64);
#pragma unroll
  for (int q = 0; q < 8; ++q) dst[q] = *(const v8h*)(src + q * 8);
}

// ---------------- elementwise kernels ----------------
__global__ void cvt_f32_f16_kernel(const float* __restrict__ in,
                                   _Float16* __restrict__ out, int n) {
  int i = blockIdx.x * 256 + threadIdx.x;
  if (i < n) out[i] = (_Float16)in[i];
}

__global__ void embed_kernel(const int* __restrict__ idx,
                             const float* __restrict__ wte,
                             float* __restrict__ x, int n) {
  int i = blockIdx.x * 256 + threadIdx.x;
  if (i < n) {
    int bt = i >> 10;  // / N_EMBD
    int c = i & (N_EMBD - 1);
    x[i] = wte[((size_t)idx[bt] << 10) + c];
  }
}

// per-head symmetric layernorm: x[bt, h*64 .. h*64+63] -> f16 out
__global__ void symln_kernel(const float* __restrict__ x,
                             const float* __restrict__ w,
                             const float* __restrict__ bsh,
                             _Float16* __restrict__ out) {
  const int gid = blockIdx.x * blockDim.x + threadIdx.x;
  if (gid >= BT * N_HEAD) return;
  const int h = gid & (N_HEAD - 1);
  const int bt = gid >> 4;
  const float* xp = x + (size_t)bt * N_EMBD + h * HEAD_DIM;
  float mu = 0.f;
#pragma unroll 8
  for (int k = 0; k < HEAD_DIM; ++k) mu += xp[k];
  mu *= (1.0f / HEAD_DIM);
  float var = 0.f;
#pragma unroll 8
  for (int k = 0; k < HEAD_DIM; ++k) {
    float d = xp[k] - mu;
    var += d * d;
  }
  var *= (1.0f / HEAD_DIM);
  const float inv = rsqrtf(var + 1e-5f);
  const float wv = w[h], bv = bsh[h];
  _Float16* op = out + (size_t)bt * N_EMBD + h * HEAD_DIM;
#pragma unroll 8
  for (int k = 0; k < HEAD_DIM; ++k)
    op[k] = (_Float16)((xp[k] - mu) * inv * wv + bv);
}

// ---------------- generic WMMA GEMM: out = A(MxK,f16) * Bw(NxK,f16)^T ----------------
// +bias (per col) +optional f32 residual; writes f32 and/or f16 output.
// block = 4 waves, wave tile = 16x64 (A fragment reused over 4 B fragments).
__global__ __launch_bounds__(128) void gemm_wmma_kernel(
    const _Float16* __restrict__ A, int lda, const _Float16* __restrict__ Bw,
    int ldb, const float* __restrict__ bias, const float* __restrict__ resid,
    float* __restrict__ outf, _Float16* __restrict__ outh, int M, int N, int K) {
  const int wave = threadIdx.x >> 5, lane = threadIdx.x & 31;
  const int hi = lane >> 4, nn = lane & 15;
  const int m0 = blockIdx.y * 64 + wave * 16;
  const int n0 = blockIdx.x * 64;
  if (m0 >= M) return;

  v8f acc[4];
#pragma unroll
  for (int j = 0; j < 4; ++j) {
#pragma unroll
    for (int r = 0; r < 8; ++r) acc[j][r] = 0.0f;
  }

  for (int k0 = 0; k0 < K; k0 += 32) {
    // Issue all fragment loads first so they pipeline in one clause,
    // then run the four WMMAs back-to-back.
    v16h a = frag_ld_g(A, lda, m0, k0);
    v16h b0 = frag_ld_g(Bw, ldb, n0, k0);
    v16h b1 = frag_ld_g(Bw, ldb, n0 + 16, k0);
    v16h b2 = frag_ld_g(Bw, ldb, n0 + 32, k0);
    v16h b3 = frag_ld_g(Bw, ldb, n0 + 48, k0);
    acc[0] = wmma16(a, b0, acc[0]);
    acc[1] = wmma16(a, b1, acc[1]);
    acc[2] = wmma16(a, b2, acc[2]);
    acc[3] = wmma16(a, b3, acc[3]);
    if (k0 + 32 < K) {  // gfx1250 global_prefetch_b8 for next K tile
      __builtin_prefetch(A + (size_t)(m0 + nn) * lda + k0 + 32, 0, 1);
      __builtin_prefetch(Bw + (size_t)(n0 + nn) * ldb + k0 + 32, 0, 1);
    }
  }

#pragma unroll
  for (int j = 0; j < 4; ++j) {
    const int col = n0 + j * 16 + nn;
    const float bj = bias ? bias[col] : 0.f;
#pragma unroll
    for (int r = 0; r < 8; ++r) {
      const int row = m0 + r + 8 * hi;
      const size_t o = (size_t)row * N + col;
      float v = acc[j][r] + bj;
      if (resid) v += resid[o];
      if (outf) outf[o] = v;
      if (outh) outh[o] = (_Float16)v;
    }
  }
}

// ---------------- flash attention with ALiBi (WMMA, online softmax) ----------------
// grid = (T/64, B*H), block = 128 (4 independent waves, 16 query rows each).
__global__ __launch_bounds__(128) void attn_flash_kernel(
    const _Float16* __restrict__ qkv, _Float16* __restrict__ yout) {
  __shared__ _Float16 lp[4][16 * 32];  // P tile (C-layout -> row major)
  __shared__ _Float16 lv[4][32 * 64];  // V tile, filled by TDM
  const int wave = threadIdx.x >> 5, lane = threadIdx.x & 31;
  const int hi = lane >> 4, nn = lane & 15;
  const int h = blockIdx.y & (N_HEAD - 1), b = blockIdx.y >> 4;
  const int t0 = (blockIdx.x * 4 + wave) * 16;

  const _Float16* Q = qkv + (size_t)b * TT * QKV_LD + h * HEAD_DIM;
  const _Float16* Kb = Q + N_EMBD;
  const _Float16* Vb = Q + 2 * N_EMBD;
  const v16h aq0 = frag_ld_g(Q, QKV_LD, t0, 0);
  const v16h aq1 = frag_ld_g(Q, QKV_LD, t0, 32);

  const float slope = exp2f(-0.5f * (float)(h + 1));  // ALiBi, 16 heads
  const float scale = 0.125f;                          // 1/sqrt(64)
  const v8f z8 = {0.f, 0.f, 0.f, 0.f, 0.f, 0.f, 0.f, 0.f};

  v8f acc[4] = {z8, z8, z8, z8};
  float mrun[8], lrun[8];
#pragma unroll
  for (int r = 0; r < 8; ++r) { mrun[r] = NEG_BIG; lrun[r] = 0.f; }

  _Float16* Pl = lp[wave];
  _Float16* Vl = lv[wave];
  const unsigned vl_lds = (unsigned)(uintptr_t)Vl;

  for (int s0 = 0; s0 < t0 + 16; s0 += 32) {
    // Start the V-tile DMA first: TDM streams global->LDS while the QK^T
    // WMMAs run.  Wait for prior-iteration LDS reads before overwriting.
    asm volatile("s_wait_dscnt 0" ::: "memory");
    if (HAVE_TDM) {
      tdm_load_tile_f16(vl_lds, Vb + (size_t)s0 * QKV_LD, (unsigned)(TT - s0),
                        QKV_LD);
    } else {
      stage_tile_vgpr(Vl, Vb + (size_t)s0 * QKV_LD, TT - s0, QKV_LD);
    }

    // S = Q * K^T  (two 16x16 subtiles over the 32 keys, K-dim 64 = 2 steps)
    v16h bk = frag_ld_g_clamp(Kb, QKV_LD, s0, 0, TT - 1);
    v8f sa = wmma16(aq0, bk, z8);
    bk = frag_ld_g_clamp(Kb, QKV_LD, s0, 32, TT - 1);
    sa = wmma16(aq1, bk, sa);
    bk = frag_ld_g_clamp(Kb, QKV_LD, s0 + 16, 0, TT - 1);
    v8f sb = wmma16(aq0, bk, z8);
    bk = frag_ld_g_clamp(Kb, QKV_LD, s0 + 16, 32, TT - 1);
    sb = wmma16(aq1, bk, sb);

    float alpha[8];
#pragma unroll
    for (int r = 0; r < 8; ++r) {
      const int t = t0 + r + 8 * hi;
      const int s1 = s0 + nn, s2 = s0 + 16 + nn;
      float a = (s1 <= t) ? sa[r] * scale + slope * (float)(s1 - t) : NEG_BIG;
      float c = (s2 <= t) ? sb[r] * scale + slope * (float)(s2 - t) : NEG_BIG;
      float mx = fmaxf(a, c);
      mx = fmaxf(mx, __shfl_xor(mx, 1, 32));
      mx = fmaxf(mx, __shfl_xor(mx, 2, 32));
      mx = fmaxf(mx, __shfl_xor(mx, 4, 32));
      mx = fmaxf(mx, __shfl_xor(mx, 8, 32));
      const float mnew = fmaxf(mrun[r], mx);
      alpha[r] = __expf(mrun[r] - mnew);
      const float pa = __expf(a - mnew);
      const float pb = __expf(c - mnew);
      sa[r] = pa;
      sb[r] = pb;
      float rs = pa + pb;
      rs += __shfl_xor(rs, 1, 32);
      rs += __shfl_xor(rs, 2, 32);
      rs += __shfl_xor(rs, 4, 32);
      rs += __shfl_xor(rs, 8, 32);
      lrun[r] = lrun[r] * alpha[r] + rs;
      mrun[r] = mnew;
    }
#pragma unroll
    for (int j = 0; j < 4; ++j) {
#pragma unroll
      for (int r = 0; r < 8; ++r) acc[j][r] *= alpha[r];
    }

    // P (C layout) -> LDS row-major, then reload in A-fragment layout
#pragma unroll
    for (int r = 0; r < 8; ++r) {
      const int row = r + 8 * hi;
      Pl[row * 32 + nn] = (_Float16)sa[r];
      Pl[row * 32 + 16 + nn] = (_Float16)sb[r];
    }
    asm volatile("s_wait_dscnt 0" ::: "memory");
    const v16h ap = frag_pack(Pl + nn * 32 + 8 * hi);

#if HAVE_TDM
    __builtin_amdgcn_s_wait_tensorcnt(0);  // V tile resident in LDS
#endif
#pragma unroll
    for (int j = 0; j < 4; ++j) {
      v16h bv;
#pragma unroll
      for (int e = 0; e < 16; ++e) {
        const int k = ((e >> 3) << 4) + 8 * hi + (e & 7);
        bv[e] = Vl[k * 64 + j * 16 + nn];
      }
      acc[j] = wmma16(ap, bv, acc[j]);
    }
  }

#pragma unroll
  for (int r = 0; r < 8; ++r) {
    const float inv = 1.0f / lrun[r];
    const int t = t0 + r + 8 * hi;
    _Float16* op = yout + ((size_t)b * TT + t) * N_EMBD + h * HEAD_DIM;
#pragma unroll
    for (int j = 0; j < 4; ++j) op[j * 16 + nn] = (_Float16)(acc[j][r] * inv);
  }
}

// ---------------- vocab-FFN: softmax(f . vc^T / t) . vc, accumulated into x ----------------
// Same flash machinery, keys/values = wte head slice (4096 rows), no mask.
// grid = (BT/64, H), block = 128.
__global__ __launch_bounds__(128) void vocab_ffn_kernel(
    const _Float16* __restrict__ fh, const _Float16* __restrict__ wteh,
    const float* __restrict__ temps_l, float* __restrict__ x) {
  __shared__ _Float16 lp[4][16 * 32];
  __shared__ _Float16 lv[4][32 * 64];
  const int wave = threadIdx.x >> 5, lane = threadIdx.x & 31;
  const int hi = lane >> 4, nn = lane & 15;
  const int h = blockIdx.y;
  const int m0 = blockIdx.x * 64 + wave * 16;

  const _Float16* Q = fh + h * HEAD_DIM;     // ld = N_EMBD, rows = bt
  const _Float16* KV = wteh + h * HEAD_DIM;  // ld = N_EMBD, rows = vocab
  const v16h aq0 = frag_ld_g(Q, N_EMBD, m0, 0);
  const v16h aq1 = frag_ld_g(Q, N_EMBD, m0, 32);
  const float invt = 1.0f / fmaxf(temps_l[h], 0.1f);
  const v8f z8 = {0.f, 0.f, 0.f, 0.f, 0.f, 0.f, 0.f, 0.f};

  v8f acc[4] = {z8, z8, z8, z8};
  float mrun[8], lrun[8];
#pragma unroll
  for (int r = 0; r < 8; ++r) { mrun[r] = NEG_BIG; lrun[r] = 0.f; }

  _Float16* Pl = lp[wave];
  _Float16* Vl = lv[wave];
  const unsigned vl_lds = (unsigned)(uintptr_t)Vl;

  for (int s0 = 0; s0 < VOCAB; s0 += 32) {
    asm volatile("s_wait_dscnt 0" ::: "memory");
    if (HAVE_TDM) {
      tdm_load_tile_f16(vl_lds, KV + (size_t)s0 * N_EMBD,
                        (unsigned)(VOCAB - s0), N_EMBD);
    } else {
      stage_tile_vgpr(Vl, KV + (size_t)s0 * N_EMBD, VOCAB - s0, N_EMBD);
    }

    v16h bk = frag_ld_g(KV, N_EMBD, s0, 0);
    v8f sa = wmma16(aq0, bk, z8);
    bk = frag_ld_g(KV, N_EMBD, s0, 32);
    sa = wmma16(aq1, bk, sa);
    bk = frag_ld_g(KV, N_EMBD, s0 + 16, 0);
    v8f sb = wmma16(aq0, bk, z8);
    bk = frag_ld_g(KV, N_EMBD, s0 + 16, 32);
    sb = wmma16(aq1, bk, sb);

    float alpha[8];
#pragma unroll
    for (int r = 0; r < 8; ++r) {
      const float a = sa[r] * invt;
      const float c = sb[r] * invt;
      float mx = fmaxf(a, c);
      mx = fmaxf(mx, __shfl_xor(mx, 1, 32));
      mx = fmaxf(mx, __shfl_xor(mx, 2, 32));
      mx = fmaxf(mx, __shfl_xor(mx, 4, 32));
      mx = fmaxf(mx, __shfl_xor(mx, 8, 32));
      const float mnew = fmaxf(mrun[r], mx);
      alpha[r] = __expf(mrun[r] - mnew);
      const float pa = __expf(a - mnew);
      const float pb = __expf(c - mnew);
      sa[r] = pa;
      sb[r] = pb;
      float rs = pa + pb;
      rs += __shfl_xor(rs, 1, 32);
      rs += __shfl_xor(rs, 2, 32);
      rs += __shfl_xor(rs, 4, 32);
      rs += __shfl_xor(rs, 8, 32);
      lrun[r] = lrun[r] * alpha[r] + rs;
      mrun[r] = mnew;
    }
#pragma unroll
    for (int j = 0; j < 4; ++j) {
#pragma unroll
      for (int r = 0; r < 8; ++r) acc[j][r] *= alpha[r];
    }

#pragma unroll
    for (int r = 0; r < 8; ++r) {
      const int row = r + 8 * hi;
      Pl[row * 32 + nn] = (_Float16)sa[r];
      Pl[row * 32 + 16 + nn] = (_Float16)sb[r];
    }
    asm volatile("s_wait_dscnt 0" ::: "memory");
    const v16h ap = frag_pack(Pl + nn * 32 + 8 * hi);

#if HAVE_TDM
    __builtin_amdgcn_s_wait_tensorcnt(0);
#endif
#pragma unroll
    for (int j = 0; j < 4; ++j) {
      v16h bv;
#pragma unroll
      for (int e = 0; e < 16; ++e) {
        const int k = ((e >> 3) << 4) + 8 * hi + (e & 7);
        bv[e] = Vl[k * 64 + j * 16 + nn];
      }
      acc[j] = wmma16(ap, bv, acc[j]);
    }
  }

  // residual accumulate into x (each wave owns its 16x64 slice exclusively)
#pragma unroll
  for (int r = 0; r < 8; ++r) {
    const float inv = 1.0f / lrun[r];
    const int row = m0 + r + 8 * hi;
    float* op = x + (size_t)row * N_EMBD + h * HEAD_DIM;
#pragma unroll
    for (int j = 0; j < 4; ++j) op[j * 16 + nn] += acc[j][r] * inv;
  }
}

// ---------------- host orchestration ----------------
extern "C" void kernel_launch(void* const* d_in, const int* in_sizes, int n_in,
                              void* d_out, int out_size, void* d_ws,
                              size_t ws_size, hipStream_t stream) {
  (void)in_sizes; (void)n_in; (void)out_size; (void)ws_size;
  const int* idx = (const int*)d_in[0];
  const float* wte = (const float*)d_in[1];
  const float* caw = (const float*)d_in[2];
  const float* cab = (const float*)d_in[3];
  const float* cpw = (const float*)d_in[4];
  const float* cpb = (const float*)d_in[5];
  const float* fw = (const float*)d_in[6];
  const float* fb = (const float*)d_in[7];
  const float* temps = (const float*)d_in[8];
  const float* ln1w = (const float*)d_in[9];
  const float* ln1b = (const float*)d_in[10];
  const float* ln2w = (const float*)d_in[11];
  const float* ln2b = (const float*)d_in[12];
  const float* lnfw = (const float*)d_in[13];
  const float* lnfb = (const float*)d_in[14];

  size_t off = 0;
  auto alloc = [&](size_t bytes) -> void* {
    off = (off + 255) & ~(size_t)255;
    void* p = (char*)d_ws + off;
    off += bytes;
    return p;
  };
  _Float16* wte_h  = (_Float16*)alloc((size_t)VOCAB * N_EMBD * 2);
  _Float16* attw_h = (_Float16*)alloc((size_t)N_LAYER * 3 * N_EMBD * N_EMBD * 2);
  _Float16* prjw_h = (_Float16*)alloc((size_t)N_LAYER * N_EMBD * N_EMBD * 2);
  _Float16* ffnw_h = (_Float16*)alloc((size_t)N_LAYER * HEAD_DIM * HEAD_DIM * 2);
  float*    xbuf   = (float*)alloc((size_t)BT * N_EMBD * 4);
  _Float16* ln_h   = (_Float16*)alloc((size_t)BT * N_EMBD * 2);
  _Float16* qkv_h  = (_Float16*)alloc((size_t)BT * 3 * N_EMBD * 2);
  _Float16* y_h    = (_Float16*)alloc((size_t)BT * N_EMBD * 2);
  _Float16* f_h    = (_Float16*)alloc((size_t)BT * N_EMBD * 2);

  auto cvt = [&](const float* src, _Float16* dst, size_t n) {
    cvt_f32_f16_kernel<<<(unsigned)((n + 255) / 256), 256, 0, stream>>>(src, dst,
                                                                        (int)n);
  };
  cvt(wte, wte_h, (size_t)VOCAB * N_EMBD);
  cvt(caw, attw_h, (size_t)N_LAYER * 3 * N_EMBD * N_EMBD);
  cvt(cpw, prjw_h, (size_t)N_LAYER * N_EMBD * N_EMBD);
  cvt(fw, ffnw_h, (size_t)N_LAYER * HEAD_DIM * HEAD_DIM);

  embed_kernel<<<(BT * N_EMBD) / 256, 256, 0, stream>>>(idx, wte, xbuf,
                                                        BT * N_EMBD);

  for (int l = 0; l < N_LAYER; ++l) {
    // ---- attention sublayer ----
    symln_kernel<<<(BT * N_HEAD) / 256, 256, 0, stream>>>(
        xbuf, ln1w + l * N_HEAD, ln1b + l * N_HEAD, ln_h);
    gemm_wmma_kernel<<<dim3(3 * N_EMBD / 64, BT / 64), 128, 0, stream>>>(
        ln_h, N_EMBD, attw_h + (size_t)l * 3 * N_EMBD * N_EMBD, N_EMBD,
        cab + (size_t)l * 3 * N_EMBD, nullptr, nullptr, qkv_h, BT, 3 * N_EMBD,
        N_EMBD);
    attn_flash_kernel<<<dim3(TT / 64, BB * N_HEAD), 128, 0, stream>>>(qkv_h,
                                                                      y_h);
    gemm_wmma_kernel<<<dim3(N_EMBD / 64, BT / 64), 128, 0, stream>>>(
        y_h, N_EMBD, prjw_h + (size_t)l * N_EMBD * N_EMBD, N_EMBD,
        cpb + (size_t)l * N_EMBD, xbuf, xbuf, nullptr, BT, N_EMBD, N_EMBD);

    // ---- vocab-FFN sublayer ----
    symln_kernel<<<(BT * N_HEAD) / 256, 256, 0, stream>>>(
        xbuf, ln2w + l * N_HEAD, ln2b + l * N_HEAD, ln_h);
    // f = ln2(x) per head @ fw^T + fb : view ln_h as [BT*H, 64]
    gemm_wmma_kernel<<<dim3(1, (BT * N_HEAD) / 64), 128, 0, stream>>>(
        ln_h, HEAD_DIM, ffnw_h + (size_t)l * HEAD_DIM * HEAD_DIM, HEAD_DIM,
        fb + (size_t)l * HEAD_DIM, nullptr, nullptr, f_h, BT * N_HEAD, HEAD_DIM,
        HEAD_DIM);
    vocab_ffn_kernel<<<dim3(BT / 64, N_HEAD), 128, 0, stream>>>(
        f_h, wte_h, temps + l * N_HEAD, xbuf);
  }

  // ---- final LN + logits ----
  symln_kernel<<<(BT * N_HEAD) / 256, 256, 0, stream>>>(xbuf, lnfw, lnfb, ln_h);
  gemm_wmma_kernel<<<dim3(VOCAB / 64, BT / 64), 128, 0, stream>>>(
      ln_h, N_EMBD, wte_h, N_EMBD, nullptr, nullptr, (float*)d_out, nullptr, BT,
      VOCAB, N_EMBD);
}
